// HeteroTimingMPNN_45896020525885
// MI455X (gfx1250) — matrix operand order repaired and checked by reading, not compiled, not run on hardware
//
#include <hip/hip_runtime.h>
#include <hip/hip_bf16.h>

// ---------------------------------------------------------------------------
// HeteroTimingMPNN for MI455X (gfx1250).
// bf16 WMMA (v_wmma_f32_16x16x32_bf16) for edge/node MLP GEMMs, fp32
// accumulate + fp32 master h for residual/LayerNorm, atomic f32 scatter-add
// for segment_sum, sortable-uint atomicMax for segment_max.
// Edge kernel: 64-edge tiles, async global->LDS gather staging (ASYNCcnt),
// register-blocked 4xM WMMA per B-fragment to cut L2 weight traffic 4x.
// ---------------------------------------------------------------------------

static constexpr int NN = 100000;  // nodes
static constexpr int EE = 400000;  // edges per type
static constexpr int HH = 128;     // hidden
static constexpr int LL = 6;       // layers
static constexpr int KK = 4;       // edge types
static constexpr int GG = 16;      // graphs

typedef __attribute__((ext_vector_type(16))) __bf16 v16bf;
typedef __attribute__((ext_vector_type(8)))  __bf16 v8bf;
typedef __attribute__((ext_vector_type(8)))  float  v8f;

__device__ __forceinline__ v16bf frag2(const __bf16* p0, const __bf16* p1) {
  v8bf a = *(const v8bf*)p0;
  v8bf b = *(const v8bf*)p1;
  return __builtin_shufflevector(a, b, 0,1,2,3,4,5,6,7,8,9,10,11,12,13,14,15);
}

__device__ __forceinline__ v8f wmma_bf16(v16bf a, v16bf b, v8f c) {
  // D(f32 16x16) = A(16x32 bf16) * B(32x16 bf16) + C
  return __builtin_amdgcn_wmma_f32_16x16x32_bf16(false, a, false, b,
                                                 (short)0, c, false, false);
}

// async 16B global -> LDS copy (GLOBAL_LOAD_ASYNC_TO_LDS_B128, ASYNCcnt)
__device__ __forceinline__ void async_copy16(unsigned lds_off, const void* gsrc) {
  asm volatile("global_load_async_to_lds_b128 %0, %1, off"
               :: "v"(lds_off), "v"((unsigned long long)(size_t)gsrc)
               : "memory");
}

__device__ __forceinline__ void async_wait0() {
  asm volatile("s_wait_asynccnt 0x0" ::: "memory");
}

// ---------------------------------------------------------------------------
// utility kernels
// ---------------------------------------------------------------------------
__global__ void k_fill_zero_f32(float* __restrict__ p, long n) {
  long i = (long)blockIdx.x * blockDim.x + threadIdx.x;
  if (i < n) p[i] = 0.f;
}

__global__ void k_fill_zero_u32(unsigned* __restrict__ p, long n) {
  long i = (long)blockIdx.x * blockDim.x + threadIdx.x;
  if (i < n) p[i] = 0u;
}

// src [B][R][C] f32 -> dst [B][C][R] bf16 (so each output column is a
// contiguous K-run: exactly the per-lane B-fragment layout for WMMA)
__global__ void k_transpose_bf16(const float* __restrict__ src,
                                 __bf16* __restrict__ dst,
                                 int R, int C, long total) {
  long i = (long)blockIdx.x * blockDim.x + threadIdx.x;
  if (i >= total) return;
  long rc = (long)R * C;
  long b = i / rc;
  long rem = i - b * rc;
  int r = (int)(rem / C);
  int c = (int)(rem - (long)r * C);
  dst[b * rc + (long)c * R + r] = (__bf16)src[i];
}

// ---------------------------------------------------------------------------
// encoders (tiny FLOPs -> VALU is fine)
// ---------------------------------------------------------------------------
__global__ void __launch_bounds__(128) k_encode_nodes(
    const float* __restrict__ x,
    const float* __restrict__ W1, const float* __restrict__ b1,
    const float* __restrict__ W2, const float* __restrict__ b2,
    float* __restrict__ h, __bf16* __restrict__ h_bf) {
  __shared__ float xin[14];
  __shared__ float hid[HH];
  const int tid = threadIdx.x;
  const size_t n = blockIdx.x;
  if (tid < 14) xin[tid] = x[n * 14 + tid];
  __syncthreads();
  float a = b1[tid];
  #pragma unroll
  for (int i = 0; i < 14; ++i) a += xin[i] * W1[i * HH + tid];
  hid[tid] = a > 0.f ? a : 0.f;
  __syncthreads();
  float o = b2[tid];
  for (int i = 0; i < HH; ++i) o += hid[i] * W2[i * HH + tid];
  h[n * HH + tid] = o;
  h_bf[n * HH + tid] = (__bf16)o;
}

__global__ void __launch_bounds__(128) k_encode_edges(
    const float* __restrict__ ea,
    const float* __restrict__ W1, const float* __restrict__ b1,
    const float* __restrict__ W2, const float* __restrict__ b2,
    __bf16* __restrict__ emb) {
  __shared__ float xin[8];
  __shared__ float hid[HH];
  const int tid = threadIdx.x;
  const size_t idx = blockIdx.x;          // k*E + e
  const int k = (int)(idx / EE);
  if (tid < 8) xin[tid] = ea[idx * 8 + tid];
  __syncthreads();
  float a = b1[(size_t)k * HH + tid];
  #pragma unroll
  for (int i = 0; i < 8; ++i) a += xin[i] * W1[((size_t)k * 8 + i) * HH + tid];
  hid[tid] = a > 0.f ? a : 0.f;
  __syncthreads();
  float o = b2[(size_t)k * HH + tid];
  for (int i = 0; i < HH; ++i) o += hid[i] * W2[((size_t)k * HH + i) * HH + tid];
  emb[idx * HH + tid] = (__bf16)o;
}

// ---------------------------------------------------------------------------
// Edge-message kernel: 64-edge tile of one edge type.
// LDS: gathered A [64x384] bf16 (48KB, via async global->LDS B128) +
//      hidden [64x256] bf16 (32KB).
// GEMM1: [64,384]x[384,256]; GEMM2: [64,256]x[256,128]; each B fragment is
// reused across 4 M-tiles from registers; atomic f32 scatter-add -> agg[dst].
// ---------------------------------------------------------------------------
__global__ void __launch_bounds__(256) k_edge_msg(
    const __bf16* __restrict__ h_bf, const __bf16* __restrict__ emb,
    const int* __restrict__ eidx,
    const __bf16* __restrict__ W1t, const float* __restrict__ b1,
    const __bf16* __restrict__ W2t, const float* __restrict__ b2,
    float* __restrict__ agg) {
  __shared__ int s_src[64];
  __shared__ int s_dst[64];
  __shared__ alignas(16) __bf16 sA[64 * 384];   // 48KB
  __shared__ alignas(16) __bf16 sO1[64 * 256];  // 32KB

  const int tid  = threadIdx.x;
  const int lane = tid & 31;
  const int wave = tid >> 5;
  const int tilesPerK = EE / 64;
  const int k  = blockIdx.x / tilesPerK;
  const int e0 = (blockIdx.x % tilesPerK) * 64;

  if (tid < 64) {
    s_src[tid] = eidx[(size_t)k * 2 * EE + e0 + tid];
    s_dst[tid] = eidx[(size_t)k * 2 * EE + EE + e0 + tid];
  }
  __syncthreads();

  // Async gather of concat(h[src], h[dst], emb[k,e]) rows into LDS, 16B units.
  for (int c = tid; c < 64 * 48; c += 256) {
    int row = c / 48;
    int off = (c % 48) * 8;
    const __bf16* src;
    if (off < 128)      src = h_bf + (size_t)s_src[row] * HH + off;
    else if (off < 256) src = h_bf + (size_t)s_dst[row] * HH + (off - 128);
    else                src = emb + ((size_t)k * EE + e0 + row) * HH + (off - 256);
    unsigned lds = (unsigned)(size_t)&sA[row * 384 + off];
    async_copy16(lds, src);
  }
  async_wait0();
  __syncthreads();

  const int mlane = lane & 15;
  const int abase = (lane >= 16) ? 8 : 0;    // A-fragment K base (16-bit layout)
  const int bbase = (lane >= 16) ? 8 : 0;    // B-fragment K base
  const int mhi   = (lane >= 16) ? 8 : 0;    // C/D row group

  // GEMM1: out1[64,256]; each wave owns 2 N-tiles, 4 M-tiles per B fragment
  for (int nt = wave; nt < 16; nt += 8) {
    int nn = nt * 16 + mlane;
    const __bf16* bcol = W1t + ((size_t)k * 256 + nn) * 384 + bbase;
    v8f acc0 = {}, acc1 = {}, acc2 = {}, acc3 = {};
    #pragma unroll
    for (int kt = 0; kt < 12; ++kt) {
      v16bf bfr = frag2(bcol + kt * 32, bcol + kt * 32 + 16);
      if (kt + 1 < 12) __builtin_prefetch(bcol + (kt + 1) * 32, 0, 0);
      const int ko = kt * 32 + abase;
      const __bf16* a0 = &sA[(mlane     ) * 384 + ko];
      const __bf16* a1 = &sA[(mlane + 16) * 384 + ko];
      const __bf16* a2 = &sA[(mlane + 32) * 384 + ko];
      const __bf16* a3 = &sA[(mlane + 48) * 384 + ko];
      acc0 = wmma_bf16(frag2(a0, a0 + 16), bfr, acc0);
      acc1 = wmma_bf16(frag2(a1, a1 + 16), bfr, acc1);
      acc2 = wmma_bf16(frag2(a2, a2 + 16), bfr, acc2);
      acc3 = wmma_bf16(frag2(a3, a3 + 16), bfr, acc3);
    }
    float bias = b1[(size_t)k * 256 + nn];
    #pragma unroll
    for (int r = 0; r < 8; ++r) {
      int m = mhi + r;
      float v0 = acc0[r] + bias; v0 = v0 > 0.f ? v0 : 0.f;
      float v1 = acc1[r] + bias; v1 = v1 > 0.f ? v1 : 0.f;
      float v2 = acc2[r] + bias; v2 = v2 > 0.f ? v2 : 0.f;
      float v3 = acc3[r] + bias; v3 = v3 > 0.f ? v3 : 0.f;
      sO1[(m     ) * 256 + nn] = (__bf16)v0;
      sO1[(m + 16) * 256 + nn] = (__bf16)v1;
      sO1[(m + 32) * 256 + nn] = (__bf16)v2;
      sO1[(m + 48) * 256 + nn] = (__bf16)v3;
    }
  }
  __syncthreads();

  // GEMM2: out2[64,128]; 8 N-tiles -> one per wave, 4 M-tiles per B fragment
  {
    const int nt = wave;
    int nn = nt * 16 + mlane;
    const __bf16* bcol = W2t + ((size_t)k * 128 + nn) * 256 + bbase;
    v8f acc0 = {}, acc1 = {}, acc2 = {}, acc3 = {};
    #pragma unroll
    for (int kt = 0; kt < 8; ++kt) {
      v16bf bfr = frag2(bcol + kt * 32, bcol + kt * 32 + 16);
      const int ko = kt * 32 + abase;
      const __bf16* a0 = &sO1[(mlane     ) * 256 + ko];
      const __bf16* a1 = &sO1[(mlane + 16) * 256 + ko];
      const __bf16* a2 = &sO1[(mlane + 32) * 256 + ko];
      const __bf16* a3 = &sO1[(mlane + 48) * 256 + ko];
      acc0 = wmma_bf16(frag2(a0, a0 + 16), bfr, acc0);
      acc1 = wmma_bf16(frag2(a1, a1 + 16), bfr, acc1);
      acc2 = wmma_bf16(frag2(a2, a2 + 16), bfr, acc2);
      acc3 = wmma_bf16(frag2(a3, a3 + 16), bfr, acc3);
    }
    float bias = b2[(size_t)k * 128 + nn];
    #pragma unroll
    for (int r = 0; r < 8; ++r) {
      int m = mhi + r;
      atomicAdd(&agg[(size_t)s_dst[m     ] * HH + nn], acc0[r] + bias);
      atomicAdd(&agg[(size_t)s_dst[m + 16] * HH + nn], acc1[r] + bias);
      atomicAdd(&agg[(size_t)s_dst[m + 32] * HH + nn], acc2[r] + bias);
      atomicAdd(&agg[(size_t)s_dst[m + 48] * HH + nn], acc3[r] + bias);
    }
  }
}

// ---------------------------------------------------------------------------
// Node-update kernel: 32 nodes/block. concat(h,agg)->MLP->residual->LN.
// ---------------------------------------------------------------------------
__global__ void __launch_bounds__(256) k_node_upd(
    float* __restrict__ h, __bf16* __restrict__ h_bf,
    const float* __restrict__ agg,
    const __bf16* __restrict__ W1t, const float* __restrict__ b1,
    const __bf16* __restrict__ W2t, const float* __restrict__ b2,
    const float* __restrict__ gamma, const float* __restrict__ beta) {
  __shared__ alignas(16) __bf16 sA[32 * 256];
  __shared__ alignas(16) __bf16 sO1[32 * 256];
  __shared__ float  sOut[32 * 128];

  const int tid  = threadIdx.x;
  const int lane = tid & 31;
  const int wave = tid >> 5;
  const size_t n0 = (size_t)blockIdx.x * 32;

  // stage concat(h_bf, bf16(agg)) rows
  for (int c = tid; c < 32 * 32; c += 256) {
    int row = c >> 5;
    int off = (c & 31) * 8;
    if (off < 128) {
      *(v8bf*)&sA[row * 256 + off] = *(const v8bf*)&h_bf[(n0 + row) * HH + off];
    } else {
      const float* s = &agg[(n0 + row) * HH + off - 128];
      #pragma unroll
      for (int i = 0; i < 8; ++i) sA[row * 256 + off + i] = (__bf16)s[i];
    }
  }
  __syncthreads();

  const int mlane = lane & 15;
  const int abase = (lane >= 16) ? 8 : 0;
  const int bbase = (lane >= 16) ? 8 : 0;
  const int mhi   = (lane >= 16) ? 8 : 0;

  // GEMM1: [32,256]x[256,256]
  for (int tt = wave; tt < 32; tt += 8) {
    int mt = tt & 1, nt = tt >> 1;
    int nn = nt * 16 + mlane;
    const __bf16* arow = &sA[(mt * 16 + mlane) * 256];
    const __bf16* bcol = W1t + (size_t)nn * 256 + bbase;
    v8f acc = {};
    #pragma unroll
    for (int kt = 0; kt < 8; ++kt) {
      v16bf a   = frag2(arow + kt * 32 + abase, arow + kt * 32 + abase + 16);
      v16bf bfr = frag2(bcol + kt * 32, bcol + kt * 32 + 16);
      if (kt + 1 < 8) __builtin_prefetch(bcol + (kt + 1) * 32, 0, 0);
      acc = wmma_bf16(a, bfr, acc);
    }
    float bias = b1[nn];
    int mb = mt * 16 + mhi;
    #pragma unroll
    for (int r = 0; r < 8; ++r) {
      float v = acc[r] + bias;
      v = v > 0.f ? v : 0.f;
      sO1[(mb + r) * 256 + nn] = (__bf16)v;
    }
  }
  __syncthreads();

  // GEMM2: [32,256]x[256,128], + bias + residual -> sOut (f32)
  for (int tt = wave; tt < 16; tt += 8) {
    int mt = tt & 1, nt = tt >> 1;
    int nn = nt * 16 + mlane;
    const __bf16* arow = &sO1[(mt * 16 + mlane) * 256];
    const __bf16* bcol = W2t + (size_t)nn * 256 + bbase;
    v8f acc = {};
    #pragma unroll
    for (int kt = 0; kt < 8; ++kt) {
      v16bf a   = frag2(arow + kt * 32 + abase, arow + kt * 32 + abase + 16);
      v16bf bfr = frag2(bcol + kt * 32, bcol + kt * 32 + 16);
      acc = wmma_bf16(a, bfr, acc);
    }
    float bias = b2[nn];
    int mb = mt * 16 + mhi;
    #pragma unroll
    for (int r = 0; r < 8; ++r) {
      int m = mb + r;
      sOut[m * 128 + nn] = acc[r] + bias + h[(n0 + m) * HH + nn];
    }
  }
  __syncthreads();

  // LayerNorm per row
  if (tid < 32) {
    const float* o = &sOut[tid * 128];
    float mu = 0.f;
    for (int j = 0; j < 128; ++j) mu += o[j];
    mu *= (1.f / 128.f);
    float var = 0.f;
    for (int j = 0; j < 128; ++j) { float d = o[j] - mu; var += d * d; }
    var *= (1.f / 128.f);
    float rstd = rsqrtf(var + 1e-5f);
    size_t node = n0 + tid;
    for (int j = 0; j < 128; ++j) {
      float y = (o[j] - mu) * rstd * gamma[j] + beta[j];
      h[node * HH + j] = y;
      h_bf[node * HH + j] = (__bf16)y;
    }
  }
}

// ---------------------------------------------------------------------------
// heads
// ---------------------------------------------------------------------------
__global__ void __launch_bounds__(64) k_node_head(
    const float* __restrict__ h,
    const float* __restrict__ W1, const float* __restrict__ b1,
    const float* __restrict__ W2, const float* __restrict__ b2,
    float* __restrict__ out) {
  __shared__ float row[128];
  __shared__ float red[64];
  const int tid = threadIdx.x;
  const size_t n = blockIdx.x;
  row[tid]      = h[n * HH + tid];
  row[tid + 64] = h[n * HH + 64 + tid];
  __syncthreads();
  float a = b1[tid];
  for (int i = 0; i < 128; ++i) a += row[i] * W1[i * 64 + tid];
  a = a > 0.f ? a : 0.f;
  red[tid] = a * W2[tid];
  __syncthreads();
  for (int s = 32; s > 0; s >>= 1) {
    if (tid < s) red[tid] += red[tid + s];
    __syncthreads();
  }
  if (tid == 0) out[n] = red[0] + b2[0];
}

__device__ __forceinline__ unsigned f32_sortable(float f) {
  unsigned u = __float_as_uint(f);
  return (u & 0x80000000u) ? ~u : (u | 0x80000000u);
}

// 8 consecutive (sorted-batch) nodes per thread: running max, flush on
// graph-id change -> ~8x fewer atomics than one-atomic-per-element
__global__ void k_pool_max(const float* __restrict__ h,
                           const int* __restrict__ bvec,
                           unsigned* __restrict__ pool) {
  const size_t nblk = NN / 8;  // 12500
  size_t t = (size_t)blockIdx.x * 256 + threadIdx.x;
  if (t >= nblk * HH) return;
  int j = (int)(t & 127);
  size_t n0 = (t >> 7) * 8;
  int curg = bvec[n0];
  float m = h[n0 * HH + j];
  #pragma unroll
  for (int i = 1; i < 8; ++i) {
    size_t n = n0 + i;
    int g = bvec[n];
    float v = h[n * HH + j];
    if (g != curg) {
      atomicMax(&pool[(size_t)curg * HH + j], f32_sortable(m));
      curg = g;
      m = v;
    } else {
      m = fmaxf(m, v);
    }
  }
  atomicMax(&pool[(size_t)curg * HH + j], f32_sortable(m));
}

__global__ void __launch_bounds__(64) k_graph_head(
    const unsigned* __restrict__ pool,
    const float* __restrict__ W1, const float* __restrict__ b1,
    const float* __restrict__ W2, const float* __restrict__ b2,
    float* __restrict__ out) {
  __shared__ float row[128];
  __shared__ float red[64];
  const int tid = threadIdx.x;
  const int g = blockIdx.x;
  for (int i = tid; i < 128; i += 64) {
    unsigned u = pool[(size_t)g * HH + i];
    u = (u & 0x80000000u) ? (u & 0x7FFFFFFFu) : ~u;  // decode
    row[i] = __uint_as_float(u);
  }
  __syncthreads();
  float a = b1[tid];
  for (int i = 0; i < 128; ++i) a += row[i] * W1[i * 64 + tid];
  a = a > 0.f ? a : 0.f;
  red[tid] = a * W2[tid];
  __syncthreads();
  for (int s = 32; s > 0; s >>= 1) {
    if (tid < s) red[tid] += red[tid + s];
    __syncthreads();
  }
  if (tid == 0) out[g] = red[0] + b2[0];
}

// ---------------------------------------------------------------------------
// host orchestration
// ---------------------------------------------------------------------------
extern "C" void kernel_launch(void* const* d_in, const int* in_sizes, int n_in,
                              void* d_out, int out_size, void* d_ws, size_t ws_size,
                              hipStream_t stream) {
  (void)in_sizes; (void)n_in; (void)out_size; (void)ws_size;

  const float* x          = (const float*)d_in[0];
  const int*   edge_index = (const int*)d_in[1];
  const float* edge_attr  = (const float*)d_in[2];
  const int*   batch_vec  = (const int*)d_in[3];
  const float* ne_W1 = (const float*)d_in[4];
  const float* ne_b1 = (const float*)d_in[5];
  const float* ne_W2 = (const float*)d_in[6];
  const float* ne_b2 = (const float*)d_in[7];
  const float* ee_W1 = (const float*)d_in[8];
  const float* ee_b1 = (const float*)d_in[9];
  const float* ee_W2 = (const float*)d_in[10];
  const float* ee_b2 = (const float*)d_in[11];
  const float* em_W1 = (const float*)d_in[12];
  const float* em_b1 = (const float*)d_in[13];
  const float* em_W2 = (const float*)d_in[14];
  const float* em_b2 = (const float*)d_in[15];
  const float* nm_W1 = (const float*)d_in[16];
  const float* nm_b1 = (const float*)d_in[17];
  const float* nm_W2 = (const float*)d_in[18];
  const float* nm_b2 = (const float*)d_in[19];
  const float* ln_g  = (const float*)d_in[20];
  const float* ln_b  = (const float*)d_in[21];
  const float* reg_W1 = (const float*)d_in[22];
  const float* reg_b1 = (const float*)d_in[23];
  const float* reg_W2 = (const float*)d_in[24];
  const float* reg_b2 = (const float*)d_in[25];
  const float* gr_W1  = (const float*)d_in[26];
  const float* gr_b1  = (const float*)d_in[27];
  const float* gr_W2  = (const float*)d_in[28];
  const float* gr_b2  = (const float*)d_in[29];

  float* out = (float*)d_out;

  char* ws = (char*)d_ws;
  size_t off = 0;
  auto take = [&](size_t bytes) -> char* {
    char* p = ws + off;
    off = (off + bytes + 255) & ~(size_t)255;
    return p;
  };
  float*    h     = (float*)take((size_t)NN * HH * 4);
  __bf16*   h_bf  = (__bf16*)take((size_t)NN * HH * 2);
  float*    agg   = (float*)take((size_t)NN * HH * 4);
  __bf16*   emb   = (__bf16*)take((size_t)KK * EE * HH * 2);
  __bf16*   eW1t  = (__bf16*)take((size_t)LL * KK * 256 * 384 * 2);
  __bf16*   eW2t  = (__bf16*)take((size_t)LL * KK * 128 * 256 * 2);
  __bf16*   nW1t  = (__bf16*)take((size_t)LL * 256 * 256 * 2);
  __bf16*   nW2t  = (__bf16*)take((size_t)LL * 128 * 256 * 2);
  unsigned* pool  = (unsigned*)take((size_t)GG * HH * 4);

  // encoders
  k_encode_nodes<<<NN, 128, 0, stream>>>(x, ne_W1, ne_b1, ne_W2, ne_b2, h, h_bf);
  k_encode_edges<<<KK * EE, 128, 0, stream>>>(edge_attr, ee_W1, ee_b1, ee_W2, ee_b2, emb);

  // weight transposes -> bf16 WMMA B-layout
  {
    long t1 = (long)LL * KK * 384 * 256;
    k_transpose_bf16<<<(unsigned)((t1 + 255) / 256), 256, 0, stream>>>(em_W1, eW1t, 384, 256, t1);
    long t2 = (long)LL * KK * 256 * 128;
    k_transpose_bf16<<<(unsigned)((t2 + 255) / 256), 256, 0, stream>>>(em_W2, eW2t, 256, 128, t2);
    long t3 = (long)LL * 256 * 256;
    k_transpose_bf16<<<(unsigned)((t3 + 255) / 256), 256, 0, stream>>>(nm_W1, nW1t, 256, 256, t3);
    long t4 = (long)LL * 256 * 128;
    k_transpose_bf16<<<(unsigned)((t4 + 255) / 256), 256, 0, stream>>>(nm_W2, nW2t, 256, 128, t4);
  }

  // message-passing layers
  const long nh = (long)NN * HH;
  for (int l = 0; l < LL; ++l) {
    k_fill_zero_f32<<<(unsigned)((nh + 255) / 256), 256, 0, stream>>>(agg, nh);
    k_edge_msg<<<KK * (EE / 64), 256, 0, stream>>>(
        h_bf, emb, edge_index,
        eW1t + (size_t)l * KK * 256 * 384, em_b1 + (size_t)l * KK * 256,
        eW2t + (size_t)l * KK * 128 * 256, em_b2 + (size_t)l * KK * 128,
        agg);
    k_node_upd<<<NN / 32, 256, 0, stream>>>(
        h, h_bf, agg,
        nW1t + (size_t)l * 256 * 256, nm_b1 + (size_t)l * 256,
        nW2t + (size_t)l * 128 * 256, nm_b2 + (size_t)l * 128,
        ln_g + (size_t)l * 128, ln_b + (size_t)l * 128);
  }

  // heads
  k_node_head<<<NN, 64, 0, stream>>>(h, reg_W1, reg_b1, reg_W2, reg_b2, out);
  k_fill_zero_u32<<<(unsigned)((GG * HH + 255) / 256), 256, 0, stream>>>(pool, (long)GG * HH);
  k_pool_max<<<(unsigned)((nh / 8 + 255) / 256), 256, 0, stream>>>(h, batch_vec, pool);
  k_graph_head<<<GG, 64, 0, stream>>>(pool, gr_W1, gr_b1, gr_W2, gr_b2, out + NN);
}
